// GCN_90271622627979
// MI455X (gfx1250) — compile-verified, matrix operand order checked by reading, not run on hardware
//
#include <hip/hip_runtime.h>
#include <cstdint>
#include <cstddef>

typedef __attribute__((ext_vector_type(16))) _Float16 v16h;
typedef __attribute__((ext_vector_type(8)))  float    v8f;

#define NN      100000
#define NE      3200000
#define NG      64
#define INF     128
#define H1      256
#define H2      128
#define OUTF    2

// ---------------- zero workspace (grid-stride) ----------------
__global__ void k_zero(float* __restrict__ p, size_t n) {
    size_t i = (size_t)blockIdx.x * blockDim.x + threadIdx.x;
    size_t stride = (size_t)gridDim.x * blockDim.x;
    for (; i < n; i += stride) p[i] = 0.0f;
}

// ---------------- degree accumulation (u32 atomics into norm buffers) ----
__global__ void k_degree(const int* __restrict__ src, const int* __restrict__ dst,
                         unsigned* __restrict__ dout, unsigned* __restrict__ din) {
    int e = blockIdx.x * blockDim.x + threadIdx.x;
    if (e < NE) {
        atomicAdd(&dout[src[e]], 1u);
        atomicAdd(&din[dst[e]], 1u);
    }
}

// ---------------- convert degrees -> rsqrt(max(deg,1)) in place ----------
__global__ void k_norm(float* __restrict__ buf) {
    int i = blockIdx.x * blockDim.x + threadIdx.x;
    if (i < 2 * NN) {
        unsigned d = ((const unsigned*)buf)[i];
        float df = (float)(d == 0u ? 1u : d);
        buf[i] = rsqrtf(df);
    }
}

// ---------------- GEMM1 via WMMA: xw = (X @ W1) * norm_out ---------------
// One block = 8 waves handles a 16-row tile; each wave does two 16x16 N-tiles.
__global__ __launch_bounds__(256) void k_gemm1(const float* __restrict__ X,
                                               const float* __restrict__ W1,
                                               const float* __restrict__ norm_out,
                                               float* __restrict__ xw) {
    __shared__ _Float16 As[16 * INF]; // 4 KB f16 staging of 16x128 feature slab
    const int tid  = threadIdx.x;
    const int row0 = blockIdx.x * 16;

    #pragma unroll
    for (int j = 0; j < 8; ++j) {
        int idx = tid * 8 + j; // contiguous -> coalesced
        As[idx] = (_Float16)X[(size_t)row0 * INF + idx];
    }
    __syncthreads();

    const int wave = tid >> 5;
    const int lane = tid & 31;
    const int mrow = lane & 15;
    const int kb   = (lane >> 4) * 8;   // lanes 0-15 -> K base 0, 16-31 -> 8
    const int hi   = lane >> 4;

    // Build four K=32 A fragments per ISA 16-bit A layout (reused for both tiles)
    v16h a[4];
    #pragma unroll
    for (int ks = 0; ks < 4; ++ks) {
        int k0 = ks * 32;
        #pragma unroll
        for (int v = 0; v < 8; ++v) {
            int kk = (v < 4) ? (kb + 2 * v) : (kb + 16 + 2 * (v - 4));
            a[ks][2 * v]     = As[mrow * INF + k0 + kk];
            a[ks][2 * v + 1] = As[mrow * INF + k0 + kk + 1];
        }
    }

    const int ncol = lane & 15;
    #pragma unroll
    for (int c = 0; c < 2; ++c) {
        int n0 = (wave * 2 + c) * 16;
        v8f acc = {};
        #pragma unroll
        for (int ks = 0; ks < 4; ++ks) {
            int k0 = ks * 32;
            v16h b; // lane = column of B (row of W1^T), halves pack 2 K values
            #pragma unroll
            for (int v = 0; v < 8; ++v) {
                int kk = (v < 4) ? (kb + 2 * v) : (kb + 16 + 2 * (v - 4));
                b[2 * v]     = (_Float16)W1[(size_t)(k0 + kk)     * H1 + n0 + ncol];
                b[2 * v + 1] = (_Float16)W1[(size_t)(k0 + kk + 1) * H1 + n0 + ncol];
            }
            acc = __builtin_amdgcn_wmma_f32_16x16x32_f16(
                false, a[ks], false, b, (short)0, acc, false, false);
        }
        // C/D layout: lane holds N=ncol, VGPR r holds M = r + 8*hi
        #pragma unroll
        for (int r = 0; r < 8; ++r) {
            int row = row0 + r + 8 * hi;
            xw[(size_t)row * H1 + n0 + ncol] = acc[r] * norm_out[row];
        }
    }
}

// ---------------- SpMM scatter: agg[dst] += xw[src] (1 wave per edge) ----
__global__ __launch_bounds__(256) void k_scatter(const int* __restrict__ src,
                                                 const int* __restrict__ dst,
                                                 const float* __restrict__ xw,
                                                 float* __restrict__ agg) {
    int gwid = (int)(((size_t)blockIdx.x * blockDim.x + threadIdx.x) >> 5);
    int lane = threadIdx.x & 31;
    if (gwid < NE) {
        const float* xs = xw  + (size_t)src[gwid] * H1;
        float*       ad = agg + (size_t)dst[gwid] * H1;
        #pragma unroll
        for (int i = 0; i < H1 / 32; ++i)
            atomicAdd(&ad[lane + 32 * i], xs[lane + 32 * i]);
    }
}

// ---------------- per-graph node counts ----------------------------------
__global__ void k_gcnt(const int* __restrict__ gid, float* __restrict__ gcnt) {
    int n = blockIdx.x * blockDim.x + threadIdx.x;
    if (n < NN) atomicAdd(&gcnt[gid[n]], 1.0f);
}

// ------- h = agg*norm_in + b1, accumulate per-graph sums -----------------
__global__ void k_pool(const float* __restrict__ agg, const float* __restrict__ norm_in,
                       const float* __restrict__ b1, const int* __restrict__ gid,
                       float* __restrict__ gsum) {
    size_t t = (size_t)blockIdx.x * blockDim.x + threadIdx.x;
    if (t < (size_t)NN * H1) {
        int n = (int)(t >> 8);
        int f = (int)(t & (H1 - 1));
        float v = agg[t] * norm_in[n] + b1[f];
        atomicAdd(&gsum[(size_t)gid[n] * H1 + f], v);
    }
}

// ---------------- xmid = relu(pooled @ W2 + b2) --------------------------
__global__ void k_mlp2(const float* __restrict__ gsum, const float* __restrict__ gcnt,
                       const float* __restrict__ W2, const float* __restrict__ b2,
                       float* __restrict__ xmid) {
    int t = blockIdx.x * blockDim.x + threadIdx.x;
    if (t < NG * H2) {
        int g = t >> 7, h = t & (H2 - 1);
        float inv = 1.0f / fmaxf(gcnt[g], 1.0f);
        float acc = b2[h];
        for (int k = 0; k < H1; ++k)
            acc += (gsum[(size_t)g * H1 + k] * inv) * W2[(size_t)k * H2 + h];
        xmid[(size_t)g * H2 + h] = fmaxf(acc, 0.0f);
    }
}

// ---------------- out = sigmoid(xmid @ W3 + b3) --------------------------
__global__ void k_mlp3(const float* __restrict__ xmid, const float* __restrict__ W3,
                       const float* __restrict__ b3, float* __restrict__ out) {
    int t = blockIdx.x * blockDim.x + threadIdx.x;
    if (t < NG * OUTF) {
        int g = t >> 1, o = t & 1;
        float acc = b3[o];
        for (int k = 0; k < H2; ++k)
            acc += xmid[(size_t)g * H2 + k] * W3[(size_t)k * OUTF + o];
        out[t] = 1.0f / (1.0f + expf(-acc));
    }
}

extern "C" void kernel_launch(void* const* d_in, const int* in_sizes, int n_in,
                              void* d_out, int out_size, void* d_ws, size_t ws_size,
                              hipStream_t stream) {
    const float* features = (const float*)d_in[0];
    const int*   src      = (const int*)  d_in[1];
    const int*   dst      = (const int*)  d_in[2];
    const int*   gids     = (const int*)  d_in[3];
    const float* W1       = (const float*)d_in[4];
    const float* b1       = (const float*)d_in[5];
    const float* W2       = (const float*)d_in[6];
    const float* b2       = (const float*)d_in[7];
    const float* W3       = (const float*)d_in[8];
    const float* b3       = (const float*)d_in[9];
    float* out = (float*)d_out;

    // workspace layout (floats)
    float* ws       = (float*)d_ws;
    float* norm_out = ws;                              // NN
    float* norm_in  = ws + NN;                         // NN
    float* xw       = ws + 2 * (size_t)NN;             // NN*H1
    float* agg      = xw + (size_t)NN * H1;            // NN*H1
    float* gsum     = agg + (size_t)NN * H1;           // NG*H1
    float* gcnt     = gsum + (size_t)NG * H1;          // NG
    float* xmid     = gcnt + NG;                       // NG*H2
    size_t total_f  = 2 * (size_t)NN + 2 * (size_t)NN * H1
                      + (size_t)NG * H1 + NG + (size_t)NG * H2;

    k_zero<<<2048, 256, 0, stream>>>(ws, total_f);
    k_degree<<<(NE + 255) / 256, 256, 0, stream>>>(src, dst,
                 (unsigned*)norm_out, (unsigned*)norm_in);
    k_norm<<<(2 * NN + 255) / 256, 256, 0, stream>>>(norm_out);
    k_gemm1<<<NN / 16, 256, 0, stream>>>(features, W1, norm_out, xw);
    k_scatter<<<(NE + 7) / 8, 256, 0, stream>>>(src, dst, xw, agg);
    k_gcnt<<<(NN + 255) / 256, 256, 0, stream>>>(gids, gcnt);
    k_pool<<<(int)(((size_t)NN * H1 + 255) / 256), 256, 0, stream>>>(agg, norm_in, b1, gids, gsum);
    k_mlp2<<<(NG * H2 + 255) / 256, 256, 0, stream>>>(gsum, gcnt, W2, b2, xmid);
    k_mlp3<<<1, 128, 0, stream>>>(xmid, W3, b3, out);
}
